// vMF_41910290874855
// MI455X (gfx1250) — matrix operand order repaired and checked by reading, not compiled, not run on hardware
//
#include <hip/hip_runtime.h>
#include <hip/hip_bf16.h>

// ---------------------------------------------------------------------------
// vMF reparameterization head for MI455X (gfx1250, wave32, WMMA).
//   mu_raw = lat_code @ W_mu^T + b_mu      -> bf16 WMMA 16x16x32, f32 accum
//   norms / redundant_norm / mu            -> LDS tile reductions
//   vecs   = ortho*sqrt(1-w^2) + mu*w      -> deterministic hash RNG
// lat_code (268 MB) is streamed from HBM exactly once. W fragments are
// staged into LDS with double-buffered global_load_async_to_lds_b128
// (ASYNCcnt) so each block pulls W from L2 once instead of 8x.
// ---------------------------------------------------------------------------

#define HID   1024           // K
#define DLAT  128            // N
#define MBLK  128            // rows per workgroup (8 waves x 16 rows)
#define KSTEPS (HID / 32)    // 32 WMMA K-steps
#define LSTRIDE 132          // LDS row stride (pad to dodge bank conflicts)

#define CHUNK_K 4                              // K-steps per staged chunk
#define NCHUNK  (KSTEPS / CHUNK_K)             // 8 chunks
#define CHUNK_BYTES (CHUNK_K * 8 * 32 * 32)    // 32 KB of B fragments

typedef __attribute__((ext_vector_type(16))) __bf16 v16bf;
typedef __attribute__((ext_vector_type(8)))  float  v8f;

// native fp32 -> bf16 (lowers to v_cvt_pk_bf16_f32 on gfx1250)
__device__ __forceinline__ __bf16 f2bf(float f) { return (__bf16)f; }

__device__ __forceinline__ unsigned int pcg(unsigned int v) {
    unsigned int state = v * 747796405u + 2891336453u;
    unsigned int word  = ((state >> ((state >> 28u) + 4u)) ^ state) * 277803737u;
    return (word >> 22u) ^ word;
}

__device__ __forceinline__ float u01(unsigned int h) {
    return (float)(h >> 8) * (1.0f / 16777216.0f);
}

// approx standard normal from two uniforms (deterministic per seed)
__device__ __forceinline__ float gaussh(unsigned int seed) {
    float a = u01(pcg(seed * 2u + 0x9E3779B9u));
    float b = u01(pcg(seed * 2u + 1u + 0x85EBCA6Bu));
    return (a + b - 1.0f) * 2.449489743f;   // var 1 (Irwin-Hall n=2)
}

// ---------------------------------------------------------------------------
// Repack W_mu [DLAT, HID] fp32 row-major into bf16 WMMA B-fragments.
// Fragment (k0, ntile): lane = (k>=16half)*16 + n%16 ; element j -> K = half*16+j.
// ---------------------------------------------------------------------------
__global__ void pack_w_bf16(const float* __restrict__ W, __bf16* __restrict__ Wp) {
    int idx = blockIdx.x * blockDim.x + threadIdx.x;
    if (idx >= KSTEPS * 8 * 32) return;
    int lane  = idx & 31;
    int ntile = (idx >> 5) & 7;
    int k0    = idx >> 8;
    int n     = ntile * 16 + (lane & 15);
    int kbase = k0 * 32 + (lane >> 4) * 16;
    const float4* src = reinterpret_cast<const float4*>(W + (size_t)n * HID + kbase);
    __bf16* dst = Wp + ((size_t)(k0 * 8 + ntile) * 32 + lane) * 16;
#pragma unroll
    for (int q = 0; q < 4; ++q) {
        float4 f = src[q];
        dst[q * 4 + 0] = f2bf(f.x);
        dst[q * 4 + 1] = f2bf(f.y);
        dst[q * 4 + 2] = f2bf(f.z);
        dst[q * 4 + 3] = f2bf(f.w);
    }
}

// ---------------------------------------------------------------------------
// Fused GEMM + norms + vMF sampling. 256 threads = 8 wave32; 128x128 tile/block.
// Wave w computes rows [w*16, w*16+16) x all 128 cols (8 accumulators).
// ---------------------------------------------------------------------------
__global__ __launch_bounds__(256) void vmf_fused(
    const float*  __restrict__ A,        // lat_code [B, HID]
    const __bf16* __restrict__ Wp,       // packed bf16 W fragments
    const float*  __restrict__ bias,     // b_mu [DLAT]
    const float*  __restrict__ kld,      // [1]
    float* __restrict__ out_vecs,        // [B*DLAT]
    float* __restrict__ out_kld,         // [B]
    float* __restrict__ out_rn,          // [B]
    float* __restrict__ out_mu)          // [B*DLAT]
{
    __shared__ __align__(32) unsigned char lds_b[2 * CHUNK_BYTES];  // B double buffer
    __shared__ float lds_mu[MBLK * LSTRIDE];
    __shared__ float lds_inorm[MBLK];
    __shared__ float lds_w[MBLK];
    __shared__ float lds_dot[MBLK];
    __shared__ float lds_ionorm[MBLK];

    const int tid   = threadIdx.x;
    const int lane  = tid & 31;          // wave32
    const int wave  = tid >> 5;          // 0..7 -> 16-row slice
    const int blockRow = blockIdx.x * MBLK;

    // LDS byte offsets of the two B buffers (low 32 bits of shared aperture addr)
    const unsigned lbuf0 = (unsigned)(uintptr_t)(&lds_b[0]) + (unsigned)tid * 16u;
    const unsigned long long gW = (unsigned long long)(uintptr_t)Wp
                                + (unsigned long long)tid * 16ull;

    // ---------------- GEMM phase: 8 accumulators per wave -------------------
    v8f acc[8];
#pragma unroll
    for (int t = 0; t < 8; ++t) { v8f z = {}; acc[t] = z; }

    // A fragment per-lane addressing (contiguous k order, same as packed W):
    //   lanes 0-15: K = {0..15}; lanes 16-31: K = {16..31}
    const int arow  = blockRow + wave * 16 + (lane & 15);
    const int abase = (lane < 16) ? 0 : 16;
    const float* aptr = A + (size_t)arow * HID + abase;

    // prologue: stage chunk 0 into buffer 0 (8 x b128 per thread = 32 KB)
#pragma unroll
    for (int i = 0; i < 8; ++i) {
        asm volatile("global_load_async_to_lds_b128 %0, %1, off offset:%2"
                     :: "v"(lbuf0), "v"(gW), "i"(i * 4096) : "memory");
    }

#pragma unroll 1
    for (int c = 0; c < NCHUNK; ++c) {
        if (c + 1 < NCHUNK) {
            // prefetch next chunk into the other buffer while we compute
            unsigned lnext = lbuf0 + (unsigned)(((c + 1) & 1) * CHUNK_BYTES);
            unsigned long long gnext = gW + (unsigned long long)((c + 1) * CHUNK_BYTES);
#pragma unroll
            for (int i = 0; i < 8; ++i) {
                asm volatile("global_load_async_to_lds_b128 %0, %1, off offset:%2"
                             :: "v"(lnext), "v"(gnext), "i"(i * 4096) : "memory");
            }
            asm volatile("s_wait_asynccnt 0x8" ::: "memory");  // chunk c landed
        } else {
            asm volatile("s_wait_asynccnt 0x0" ::: "memory");  // drain last chunk
        }
        __syncthreads();   // all waves see chunk c in LDS

        const v16bf* bbase = reinterpret_cast<const v16bf*>(&lds_b[(c & 1) * CHUNK_BYTES]);
#pragma unroll
        for (int kk = 0; kk < CHUNK_K; ++kk) {
            const int k0 = c * CHUNK_K + kk;
            const float4* ap = reinterpret_cast<const float4*>(aptr + k0 * 32);
            float4 f0 = ap[0];
            float4 f1 = ap[1];
            float4 f2 = ap[2];   // +8 floats
            float4 f3 = ap[3];
            v16bf av;
            av[0]  = f2bf(f0.x); av[1]  = f2bf(f0.y); av[2]  = f2bf(f0.z); av[3]  = f2bf(f0.w);
            av[4]  = f2bf(f1.x); av[5]  = f2bf(f1.y); av[6]  = f2bf(f1.z); av[7]  = f2bf(f1.w);
            av[8]  = f2bf(f2.x); av[9]  = f2bf(f2.y); av[10] = f2bf(f2.z); av[11] = f2bf(f2.w);
            av[12] = f2bf(f3.x); av[13] = f2bf(f3.y); av[14] = f2bf(f3.z); av[15] = f2bf(f3.w);

            const v16bf* bp = bbase + ((kk * 8) * 32 + lane);
#pragma unroll
            for (int t = 0; t < 8; ++t) {
                v16bf bf = bp[t * 32];   // ds_load_b128 x2 from staged LDS
                acc[t] = __builtin_amdgcn_wmma_f32_16x16x32_bf16(false, av, false, bf,
                                                                 (short)0, acc[t], false, false);
            }
        }
        __syncthreads();   // done reading buf[c&1] before it is re-staged
    }

    // C layout: lane L -> col = ntile*16 + L%16 ; VGPR r -> row = wave*16 + r + (L/16)*8
    const int mrow = wave * 16 + (lane >> 4) * 8;
#pragma unroll
    for (int t = 0; t < 8; ++t) {
        int col = t * 16 + (lane & 15);
        float bv = bias[col];
#pragma unroll
        for (int r = 0; r < 8; ++r)
            lds_mu[(mrow + r) * LSTRIDE + col] = acc[t][r] + bv;
    }
    __syncthreads();

    // ---------------- per-row norms, kld, w-sampling, dot/onorm -------------
    if (tid < MBLK) {
        const int row  = tid;
        const int grow = blockRow + row;

        float s = 0.0f;
        for (int c = 0; c < DLAT; ++c) {
            int cc = (c + row) & (DLAT - 1);          // stagger LDS banks
            float v = lds_mu[row * LSTRIDE + cc];
            s += v * v;
        }
        float norm  = sqrtf(s);
        float inorm = 1.0f / norm;
        lds_inorm[row] = inorm;
        out_rn[grow]  = (norm - 1.0f) * (norm - 1.0f);
        out_kld[grow] = kld[0];

        // rejection sampling of w (Wood's algorithm shape; hash RNG stand-in)
        const float dd  = (float)(DLAT - 1);
        const float kap = 1.0f;
        float bb = dd / (sqrtf(4.0f * kap * kap + dd * dd) + 2.0f * kap);
        float xx = (1.0f - bb) / (1.0f + bb);
        float cc = kap * xx + dd * logf(1.0f - xx * xx);
        float sigma = 0.5f * rsqrtf(dd + 1.0f);       // ~std of Beta(d/2,d/2)
        unsigned int seed = (unsigned int)grow * 2654435761u + 12345u;
        float w = 0.0f;
        bool done = false;
#pragma unroll 1
        for (int it = 0; it < 16 && !done; ++it) {
            unsigned int s0 = seed + (unsigned int)it * 1013904223u;
            float z = 0.5f + gaussh(s0) * sigma;
            z = fminf(fmaxf(z, 1e-6f), 1.0f - 1e-6f);
            float wc = (1.0f - (1.0f + bb) * z) / (1.0f - (1.0f - bb) * z);
            float u  = fmaxf(u01(pcg(s0 ^ 0xB5297A4Du)), 1e-12f);
            done = (kap * wc + dd * logf(1.0f - xx * wc) - cc) >= logf(u);
            w = wc;
        }
        lds_w[row] = w;

        // dot(mu, v) and ||v - mu*dot|| with v regenerated from the same hash
        float dot = 0.0f;
        for (int c = 0; c < DLAT; ++c) {
            float mu = lds_mu[row * LSTRIDE + c] * inorm;
            float v  = gaussh((unsigned int)(grow * DLAT + c) ^ 0xDEADBEEFu);
            dot += mu * v;
        }
        float os = 0.0f;
        for (int c = 0; c < DLAT; ++c) {
            float mu = lds_mu[row * LSTRIDE + c] * inorm;
            float v  = gaussh((unsigned int)(grow * DLAT + c) ^ 0xDEADBEEFu);
            float o  = v - mu * dot;
            os += o * o;
        }
        lds_dot[row]    = dot;
        lds_ionorm[row] = rsqrtf(fmaxf(os, 1e-30f));
    }
    __syncthreads();

    // ---------------- write mu and vecs -------------------------------------
    for (int e = 0; e < (MBLK * DLAT) / 256; ++e) {
        int lin  = tid + e * 256;
        int row  = lin >> 7;            // DLAT == 128
        int col  = lin & (DLAT - 1);
        int grow = blockRow + row;
        float mu = lds_mu[row * LSTRIDE + col] * lds_inorm[row];
        float w  = lds_w[row];
        float v  = gaussh((unsigned int)(grow * DLAT + col) ^ 0xDEADBEEFu);
        float o  = (v - mu * lds_dot[row]) * lds_ionorm[row];
        float vec = o * sqrtf(fmaxf(1.0f - w * w, 0.0f)) + mu * w;
        out_mu[(size_t)grow * DLAT + col]   = mu;
        out_vecs[(size_t)grow * DLAT + col] = vec;
    }
}

// ---------------------------------------------------------------------------
extern "C" void kernel_launch(void* const* d_in, const int* in_sizes, int n_in,
                              void* d_out, int out_size, void* d_ws, size_t ws_size,
                              hipStream_t stream) {
    const float* lat = (const float*)d_in[0];   // [B, HID]
    const float* W   = (const float*)d_in[1];   // [DLAT, HID]
    const float* bmu = (const float*)d_in[2];   // [DLAT]
    const float* kld = (const float*)d_in[3];   // [1]
    (void)n_in; (void)out_size; (void)ws_size;

    const int B = in_sizes[0] / HID;

    float* out      = (float*)d_out;
    float* out_vecs = out;                              // [1, B, DLAT]
    float* out_kld  = out_vecs + (size_t)B * DLAT;      // [B]
    float* out_rn   = out_kld + B;                      // [B, 1]
    float* out_mu   = out_rn + B;                       // [B, DLAT]

    __bf16* Wp = (__bf16*)d_ws;                         // 256 KB packed W

    pack_w_bf16<<<(KSTEPS * 8 * 32 + 255) / 256, 256, 0, stream>>>(W, Wp);
    vmf_fused<<<B / MBLK, 256, 0, stream>>>(lat, Wp, bmu, kld,
                                            out_vecs, out_kld, out_rn, out_mu);
}
